// JTNNEncoder_77910706750066
// MI455X (gfx1250) — matrix operand order repaired and checked by reading, not compiled, not run on hardware
//
#include <hip/hip_runtime.h>

#define HVALID 450
#define HP     512
#define NNODES 5120
#define NMESS  10241
#define NM5    (NMESS * 5)
#define GRU_DEPTH 8
#define KTILES (HP / 32)          // 16 k-steps
#define NTILES (HP / 16)          // 32 n-fragment tiles

typedef __attribute__((ext_vector_type(16))) __bf16 v16bf;
typedef __attribute__((ext_vector_type(8)))  float  v8f;

union ABfrag { v16bf v; unsigned u[8]; };

__device__ __forceinline__ unsigned short f2bf(float f) {
  union { float f; unsigned u; } x; x.f = f;
  unsigned r = x.u + 0x7FFFu + ((x.u >> 16) & 1u);   // round-to-nearest-even
  return (unsigned short)(r >> 16);
}
__device__ __forceinline__ float bf2f(unsigned short s) {
  union { unsigned u; float f; } x; x.u = ((unsigned)s) << 16;
  return x.f;
}

// ---------------------------------------------------------------------------
// WMMA GEMM: C[M, 512] = A_bf16[M(idx), 512] * B[512, 512]  (+fused epilogue)
//   A: bf16 rows, stride HP; optional gather via aIdx (row i -> A[aIdx[i]]).
//   Bp: fragment-major pack: [kt(16)][nt(32)][lane(32)][8 dwords]; each lane's
//       fragment is 32 contiguous bytes -> 2x global_load_b128, wave-coalesced.
//       dword v of (kt, nt, lane): K-pair kd = kt*16 + (lane>=16?4:0) + (v&3) + (v>=4?8:0),
//                                  n = nt*16 + (lane&15).
//   act: 0 none, 1 sigmoid, 2 tanh, 3 relu
//   If zPtr != null:  out = (1-z)*sumh + z*act(acc+cin+bias), row 0 masked to 0.
// Block: 256 threads = 8 waves. Wave tile 32x32 = 4 WMMAs / k-step.
// Block tile: 128(M) x 64(N).
// ---------------------------------------------------------------------------
__global__ __launch_bounds__(256)
void gemm_bf16_wmma(const unsigned short* __restrict__ A,
                    const int* __restrict__ aIdx,
                    const unsigned* __restrict__ Bp,
                    int M,
                    const float* __restrict__ cin, int cinDiv,
                    const float* __restrict__ bias,
                    int act,
                    const float* __restrict__ zPtr,
                    const float* __restrict__ sumhPtr,
                    float* __restrict__ outF,
                    unsigned short* __restrict__ outB,
                    int outLd, int nValid)
{
  const int lane  = threadIdx.x & 31;
  const int wave  = threadIdx.x >> 5;
  const int mBase = blockIdx.x * 128 + (wave >> 1) * 32;
  const int nBase = blockIdx.y * 64  + (wave & 1) * 32;
  const int half  = lane >> 4;       // 0: lanes 0-15, 1: lanes 16-31
  const int l16   = lane & 15;

  // Two A rows per lane (m-frag 0 and 1). Clamp for full-EXEC WMMA; stores guarded.
  int r0 = mBase + l16;        r0 = r0 < M ? r0 : (M - 1);
  int r1 = mBase + 16 + l16;   r1 = r1 < M ? r1 : (M - 1);
  int ar0 = aIdx ? aIdx[r0] : r0;
  int ar1 = aIdx ? aIdx[r1] : r1;
  const unsigned* __restrict__ Au0 = (const unsigned*)(A + (size_t)ar0 * HP);
  const unsigned* __restrict__ Au1 = (const unsigned*)(A + (size_t)ar1 * HP);

  // B fragment bases for the wave's two n-tiles
  const int nt0 = nBase >> 4;
  const unsigned* __restrict__ B0 = Bp + ((size_t)nt0 * 32 + lane) * 8;
  const unsigned* __restrict__ B1 = B0 + 32 * 8;   // nt0 + 1

  v8f acc00 = {}, acc01 = {}, acc10 = {}, acc11 = {};

#pragma unroll
  for (int kt = 0; kt < KTILES; ++kt) {
    const int kd = kt * 16;                   // dword (K-pair) base within A row
    ABfrag a0, a1;
#pragma unroll
    for (int v = 0; v < 4; ++v) {
      a0.u[v]     = Au0[kd + half * 4 + v];       // K = half*8 + 2v
      a0.u[4 + v] = Au0[kd + 8 + half * 4 + v];   // K = 16 + half*8 + 2v
      a1.u[v]     = Au1[kd + half * 4 + v];
      a1.u[4 + v] = Au1[kd + 8 + half * 4 + v];
    }
    ABfrag b0, b1;
    const unsigned* __restrict__ pb0 = B0 + (size_t)kt * (NTILES * 32 * 8);
    const unsigned* __restrict__ pb1 = B1 + (size_t)kt * (NTILES * 32 * 8);
#pragma unroll
    for (int v = 0; v < 8; ++v) { b0.u[v] = pb0[v]; b1.u[v] = pb1[v]; }

    acc00 = __builtin_amdgcn_wmma_f32_16x16x32_bf16(false, a0.v, false, b0.v, (short)0, acc00, false, false);
    acc01 = __builtin_amdgcn_wmma_f32_16x16x32_bf16(false, a0.v, false, b1.v, (short)0, acc01, false, false);
    acc10 = __builtin_amdgcn_wmma_f32_16x16x32_bf16(false, a1.v, false, b0.v, (short)0, acc10, false, false);
    acc11 = __builtin_amdgcn_wmma_f32_16x16x32_bf16(false, a1.v, false, b1.v, (short)0, acc11, false, false);
  }

  // Epilogue. Lane l: col = nBase + t*16 + l16; rows = mBase + mf*16 + half*8 + i.
#pragma unroll
  for (int mf = 0; mf < 2; ++mf) {
    v8f aN0 = mf == 0 ? acc00 : acc10;
    v8f aN1 = mf == 0 ? acc01 : acc11;
#pragma unroll
    for (int i = 0; i < 8; ++i) {
      int row = mBase + mf * 16 + half * 8 + i;
      if (row >= M) continue;
#pragma unroll
      for (int t = 0; t < 2; ++t) {
        int col  = nBase + t * 16 + l16;
        float v  = t == 0 ? aN0[i] : aN1[i];
        if (cin)  v += cin[(size_t)(row / cinDiv) * HP + col];
        if (bias) v += (col < HVALID) ? bias[col] : 0.0f;
        if      (act == 1) v = 1.0f / (1.0f + __expf(-v));
        else if (act == 2) v = tanhf(v);
        else if (act == 3) v = v > 0.0f ? v : 0.0f;
        if (zPtr) {  // fused GRU update
          float zz = zPtr[(size_t)row * HP + col];
          v = (1.0f - zz) * sumhPtr[(size_t)row * HP + col] + zz * v;
          if (row == 0) v = 0.0f;   // dummy message slot
        }
        if (col < nValid) {
          if (outF) outF[(size_t)row * outLd + col] = v;
          if (outB) outB[(size_t)row * HP + col] = f2bf(v);
        }
      }
    }
  }
}

// ---------------------------------------------------------------------------
// Elementwise / gather kernels
// ---------------------------------------------------------------------------
__global__ void fill_f32(float* p, float v, long long n) {
  long long i = (long long)blockIdx.x * 256 + threadIdx.x;
  if (i < n) p[i] = v;
}
__global__ void fill_u16(unsigned short* p, unsigned short v, long long n) {
  long long i = (long long)blockIdx.x * 256 + threadIdx.x;
  if (i < n) p[i] = v;
}

// pack weight [450,450] (row-major, ld=450) into fragment-major bf16 pack:
// flat idx = ((kt*NTILES + nt)*32 + lane)*8 + v
__global__ void pack_weight(const float* __restrict__ W, unsigned* __restrict__ out) {
  int idx = blockIdx.x * 256 + threadIdx.x;
  if (idx >= KTILES * NTILES * 32 * 8) return;
  int v    = idx & 7;
  int lane = (idx >> 3) & 31;
  int nt   = (idx >> 8) & (NTILES - 1);
  int kt   = idx >> 13;
  int kd   = kt * 16 + ((lane >> 4) ? 4 : 0) + (v & 3) + ((v >> 2) << 3);   // K-pair index
  int n    = nt * 16 + (lane & 15);
  unsigned lo = 0, hi = 0;
  if (n < HVALID) {
    int k0 = kd * 2, k1 = k0 + 1;
    if (k0 < HVALID) lo = f2bf(W[(size_t)k0 * HVALID + n]);
    if (k1 < HVALID) hi = f2bf(W[(size_t)k1 * HVALID + n]);
  }
  out[idx] = lo | (hi << 16);
}

// x_b[n][c] = bf16(emb[fnode[n]][c]) padded to 512 cols
__global__ void embed_nodes(const int* __restrict__ fnode, const float* __restrict__ emb,
                            unsigned short* __restrict__ x_b) {
  long long idx = (long long)blockIdx.x * 256 + threadIdx.x;
  if (idx >= (long long)NNODES * HP) return;
  int n = (int)(idx >> 9), c = (int)(idx & (HP - 1));
  unsigned short v = 0;
  if (c < HVALID) v = f2bf(emb[(size_t)fnode[n] * HVALID + c]);
  x_b[idx] = v;
}

// sum_h[m][c] = sum_j h[mess_graph[m][j]][c]  (f32 + bf16 copies)
__global__ void neigh_sum_h(const int* __restrict__ mg, const float* __restrict__ h_f,
                            float* __restrict__ sumh_f, unsigned short* __restrict__ sumh_b) {
  long long idx = (long long)blockIdx.x * 256 + threadIdx.x;
  if (idx >= (long long)NMESS * HP) return;
  int m = (int)(idx >> 9), c = (int)(idx & (HP - 1));
  const int* g = mg + (size_t)m * 5;
  float s = 0.0f;
#pragma unroll
  for (int j = 0; j < 5; ++j) s += h_f[(size_t)g[j] * HP + c];
  sumh_f[idx] = s;
  sumh_b[idx] = f2bf(s);
}

// sum_gh[m][c] = sum_j r[m*5+j][c] * h[mess_graph[m][j]][c]
__global__ void neigh_sum_gh(const int* __restrict__ mg, const unsigned short* __restrict__ r_b,
                             const float* __restrict__ h_f, unsigned short* __restrict__ sumgh_b) {
  long long idx = (long long)blockIdx.x * 256 + threadIdx.x;
  if (idx >= (long long)NMESS * HP) return;
  int m = (int)(idx >> 9), c = (int)(idx & (HP - 1));
  const int* g = mg + (size_t)m * 5;
  float s = 0.0f;
#pragma unroll
  for (int j = 0; j < 5; ++j)
    s += bf2f(r_b[(size_t)(m * 5 + j) * HP + c]) * h_f[(size_t)g[j] * HP + c];
  sumgh_b[idx] = f2bf(s);
}

// sum_mess[n][c] = sum_j h[node_graph[n][j]][c]  (bf16)
__global__ void node_sum_mess(const int* __restrict__ ng, const float* __restrict__ h_f,
                              unsigned short* __restrict__ summess_b) {
  long long idx = (long long)blockIdx.x * 256 + threadIdx.x;
  if (idx >= (long long)NNODES * HP) return;
  int n = (int)(idx >> 9), c = (int)(idx & (HP - 1));
  const int* g = ng + (size_t)n * 6;
  float s = 0.0f;
#pragma unroll
  for (int j = 0; j < 6; ++j) s += h_f[(size_t)g[j] * HP + c];
  summess_b[idx] = f2bf(s);
}

// ---------------------------------------------------------------------------
extern "C" void kernel_launch(void* const* d_in, const int* in_sizes, int n_in,
                              void* d_out, int out_size, void* d_ws, size_t ws_size,
                              hipStream_t stream) {
  const int*   fnode = (const int*)  d_in[0];
  const int*   fmess = (const int*)  d_in[1];
  const int*   ngr   = (const int*)  d_in[2];   // node_graph [5120,6]
  const int*   mgr   = (const int*)  d_in[3];   // mess_graph [10241,5]
  const float* emb   = (const float*)d_in[4];
  const float* Wz    = (const float*)d_in[5];   // [900,450]
  const float* bz    = (const float*)d_in[6];
  const float* Wr    = (const float*)d_in[7];   // [450,450]
  const float* Ur    = (const float*)d_in[8];   // [450,450]
  const float* bu    = (const float*)d_in[9];
  const float* Wh    = (const float*)d_in[10];  // [900,450]
  const float* bh    = (const float*)d_in[11];
  const float* Wo    = (const float*)d_in[12];  // [900,450]
  const float* bo    = (const float*)d_in[13];
  float* out = (float*)d_out;
  (void)in_sizes; (void)n_in; (void)ws_size;

  // ---- workspace carve-up ----
  char* wp = (char*)d_ws;
  auto carve = [&](size_t bytes) -> void* {
    void* p = (void*)wp;
    wp += (bytes + 255) & ~(size_t)255;
    return p;
  };
  const size_t MHP = (size_t)NMESS * HP, NHP = (size_t)NNODES * HP;
  const size_t PKW = (size_t)KTILES * NTILES * 32 * 8;   // dwords per packed weight

  unsigned short* x_b      = (unsigned short*)carve(NHP * 2);
  unsigned* pWzT = (unsigned*)carve(PKW * 4);
  unsigned* pWzB = (unsigned*)carve(PKW * 4);
  unsigned* pWr  = (unsigned*)carve(PKW * 4);
  unsigned* pUr  = (unsigned*)carve(PKW * 4);
  unsigned* pWhT = (unsigned*)carve(PKW * 4);
  unsigned* pWhB = (unsigned*)carve(PKW * 4);
  unsigned* pWoT = (unsigned*)carve(PKW * 4);
  unsigned* pWoB = (unsigned*)carve(PKW * 4);
  float* r1   = (float*)carve(MHP * 4);
  float* zx   = (float*)carve(MHP * 4);
  float* hx   = (float*)carve(MHP * 4);
  float* xwo  = (float*)carve(NHP * 4);
  float* h_f  = (float*)carve(MHP * 4);
  unsigned short* h_b     = (unsigned short*)carve(MHP * 2);
  float* sumh_f = (float*)carve(MHP * 4);
  unsigned short* sumh_b  = (unsigned short*)carve(MHP * 2);
  float* z_f  = (float*)carve(MHP * 4);
  unsigned short* r_b     = (unsigned short*)carve((size_t)NM5 * HP * 2);
  unsigned short* sumgh_b = (unsigned short*)carve(MHP * 2);
  unsigned short* summess_b = (unsigned short*)carve(NHP * 2);

  const dim3 blk(256);
  auto g1 = [](long long n) { return dim3((unsigned)((n + 255) / 256)); };
  auto gemmGrid = [](int M) { return dim3((unsigned)((M + 127) / 128), HP / 64); };

  // ---- one-time prep (re-done each call for determinism) ----
  hipLaunchKernelGGL(pack_weight, g1(PKW), blk, 0, stream, Wz,                 pWzT);
  hipLaunchKernelGGL(pack_weight, g1(PKW), blk, 0, stream, Wz + 450 * HVALID,  pWzB);
  hipLaunchKernelGGL(pack_weight, g1(PKW), blk, 0, stream, Wr,                 pWr);
  hipLaunchKernelGGL(pack_weight, g1(PKW), blk, 0, stream, Ur,                 pUr);
  hipLaunchKernelGGL(pack_weight, g1(PKW), blk, 0, stream, Wh,                 pWhT);
  hipLaunchKernelGGL(pack_weight, g1(PKW), blk, 0, stream, Wh + 450 * HVALID,  pWhB);
  hipLaunchKernelGGL(pack_weight, g1(PKW), blk, 0, stream, Wo,                 pWoT);
  hipLaunchKernelGGL(pack_weight, g1(PKW), blk, 0, stream, Wo + 450 * HVALID,  pWoB);
  hipLaunchKernelGGL(embed_nodes, g1(NHP), blk, 0, stream, fnode, emb, x_b);

  // loop-invariant GEMMs; fmess gather fused into A operand
  hipLaunchKernelGGL(gemm_bf16_wmma, gemmGrid(NMESS), blk, 0, stream,
                     x_b, fmess, pWr,  NMESS, (const float*)nullptr, 1, (const float*)nullptr, 0,
                     (const float*)nullptr, (const float*)nullptr, r1, (unsigned short*)nullptr, HP, HP);
  hipLaunchKernelGGL(gemm_bf16_wmma, gemmGrid(NMESS), blk, 0, stream,
                     x_b, fmess, pWzT, NMESS, (const float*)nullptr, 1, (const float*)nullptr, 0,
                     (const float*)nullptr, (const float*)nullptr, zx, (unsigned short*)nullptr, HP, HP);
  hipLaunchKernelGGL(gemm_bf16_wmma, gemmGrid(NMESS), blk, 0, stream,
                     x_b, fmess, pWhT, NMESS, (const float*)nullptr, 1, (const float*)nullptr, 0,
                     (const float*)nullptr, (const float*)nullptr, hx, (unsigned short*)nullptr, HP, HP);
  hipLaunchKernelGGL(gemm_bf16_wmma, gemmGrid(NNODES), blk, 0, stream,
                     x_b, (const int*)nullptr, pWoT, NNODES, (const float*)nullptr, 1, (const float*)nullptr, 0,
                     (const float*)nullptr, (const float*)nullptr, xwo, (unsigned short*)nullptr, HP, HP);

  // h0 = 0
  hipLaunchKernelGGL(fill_f32, g1((long long)MHP), blk, 0, stream, h_f, 0.0f, (long long)MHP);
  hipLaunchKernelGGL(fill_u16, g1((long long)MHP), blk, 0, stream, h_b, (unsigned short)0, (long long)MHP);

  // ---- GRU depth loop ----
  for (int d = 0; d < GRU_DEPTH; ++d) {
    hipLaunchKernelGGL(neigh_sum_h, g1((long long)MHP), blk, 0, stream, mgr, h_f, sumh_f, sumh_b);

    // z = sigmoid(zx + sum_h @ WzB + bz)
    hipLaunchKernelGGL(gemm_bf16_wmma, gemmGrid(NMESS), blk, 0, stream,
                       sumh_b, (const int*)nullptr, pWzB, NMESS, zx, 1, bz, 1,
                       (const float*)nullptr, (const float*)nullptr, z_f, (unsigned short*)nullptr, HP, HP);

    // r = sigmoid(r1 + h[mess_graph] @ Ur + bu)  (gather fused; bf16 output)
    hipLaunchKernelGGL(gemm_bf16_wmma, gemmGrid(NM5), blk, 0, stream,
                       h_b, mgr, pUr, NM5, r1, 5, bu, 1,
                       (const float*)nullptr, (const float*)nullptr, (float*)nullptr, r_b, HP, HP);

    hipLaunchKernelGGL(neigh_sum_gh, g1((long long)MHP), blk, 0, stream, mgr, r_b, h_f, sumgh_b);

    // h = ((1-z)*sum_h + z*tanh(hx + sum_gh @ WhB + bh)) * mask  (fully fused epilogue)
    hipLaunchKernelGGL(gemm_bf16_wmma, gemmGrid(NMESS), blk, 0, stream,
                       sumgh_b, (const int*)nullptr, pWhB, NMESS, hx, 1, bh, 2,
                       z_f, sumh_f, h_f, h_b, HP, HP);
  }

  // ---- output stage ----
  hipLaunchKernelGGL(node_sum_mess, g1((long long)NHP), blk, 0, stream, ngr, h_f, summess_b);

  // tree_vecs = relu(xwo + sum_mess @ WoB + bo), written straight to d_out [5120,450]
  hipLaunchKernelGGL(gemm_bf16_wmma, gemmGrid(NNODES), blk, 0, stream,
                     summess_b, (const int*)nullptr, pWoB, NNODES, xwo, 1, bo, 3,
                     (const float*)nullptr, (const float*)nullptr, out, (unsigned short*)nullptr,
                     HVALID, HVALID);

  // messages output = zeros
  long long msgCount = (long long)out_size - (long long)NNODES * HVALID;
  if (msgCount > 0)
    hipLaunchKernelGGL(fill_f32, g1(msgCount), blk, 0, stream,
                       out + (size_t)NNODES * HVALID, 0.0f, msgCount);
}
